// MoE_43885975830669
// MI455X (gfx1250) — matrix operand order repaired and checked by reading, not compiled, not run on hardware
//
#include <hip/hip_runtime.h>
#include <hip/hip_bf16.h>

// ---------------- problem constants (match reference) ----------------
constexpr int NB    = 2048;   // batch
constexpr int ND    = 1024;   // input dim
constexpr int NE    = 64;     // experts
constexpr int NK    = 4;      // top-k
constexpr int NSMAX = 1536;   // max expert width (padded-stacked)
constexpr float EPS = 1e-5f;

constexpr int SCHUNK = 128;               // s-columns per block (8 waves x 16)
constexpr int NCHUNK = NSMAX / SCHUNK;    // 12

// fragment grid geometry (bf16 WMMA 16x16x32)
constexpr int ST1 = NSMAX / 16;           // 96  s-tiles  (layer-1 N tiles)
constexpr int KC1 = ND / 32;              // 32  k-chunks (layer-1 K)
constexpr int DT2 = ND / 16;              // 64  d-tiles  (layer-2 N tiles)
constexpr int SC2 = NSMAX / 32;           // 48  s-chunks (layer-2 K)
constexpr int F1TOT = NE * ST1 * KC1;     // 196608 W1 fragments (1 KB each)
constexpr int F2TOT = NE * DT2 * SC2;     // 196608 W2 fragments (1 KB each)

// ---------------- WMMA fragment types ----------------
typedef __bf16 v16bf __attribute__((ext_vector_type(16)));
typedef float  v8f   __attribute__((ext_vector_type(8)));
typedef unsigned short u16x8 __attribute__((ext_vector_type(8)));

union Frag {
    v16bf v;
    unsigned short u[16];
    u16x8 h[2];
};

// CDNA5 async global->LDS path (ASYNCcnt-tracked), compile-safe guards.
#if defined(__AMDGCN__) && __has_builtin(__builtin_amdgcn_global_load_async_to_lds_b128)
  #define USE_ASYNC_LDS 1
  typedef int v4i_async __attribute__((ext_vector_type(4)));
  typedef __attribute__((address_space(1))) v4i_async* as1_v4i;
  typedef __attribute__((address_space(3))) v4i_async* as3_v4i;
#else
  #define USE_ASYNC_LDS 0
#endif

static __device__ __forceinline__ void wait_async_lds() {
#if USE_ASYNC_LDS
  #if __has_builtin(__builtin_amdgcn_s_wait_asynccnt)
    __builtin_amdgcn_s_wait_asynccnt(0);
  #else
    asm volatile("s_wait_asynccnt 0x0" ::: "memory");
  #endif
#endif
}

// fp32 -> bf16 round-to-nearest-even
static __device__ __forceinline__ unsigned short f2bf(float f) {
    unsigned u = __float_as_uint(f);
    u += 0x7FFFu + ((u >> 16) & 1u);
    return (unsigned short)(u >> 16);
}
static __device__ __forceinline__ unsigned pack2bf(float lo, float hi) {
    return (unsigned)f2bf(lo) | ((unsigned)f2bf(hi) << 16);
}

// =====================================================================
// Kernel 0: one-shot weight convert + swizzle into WMMA B-fragment order.
// =====================================================================
__global__ __launch_bounds__(256) void convert_weights_kernel(
    const float* __restrict__ W1, const float* __restrict__ W2,
    unsigned short* __restrict__ W1T, unsigned short* __restrict__ W2T)
{
    const int lane = threadIdx.x & 31;
    const int fid  = blockIdx.x * 8 + (threadIdx.x >> 5);
    const int ncol = lane & 15;
    const int hsel = lane >> 4;

    union { unsigned short u[16]; uint4 q[2]; } o;
    unsigned short* dst;

    if (fid < F1TOT) {
        // W1 fragment: f = (e*ST1 + st)*KC1 + kc ; K = d, N = s
        const int kc = fid % KC1;
        const int st = (fid / KC1) % ST1;
        const int e  = fid / (KC1 * ST1);
        const float* src = W1 + ((size_t)e * ND + kc * 32 + hsel * 16) * NSMAX
                              + st * 16 + ncol;
        #pragma unroll
        for (int j = 0; j < 16; ++j) o.u[j] = f2bf(src[(size_t)j * NSMAX]);
        dst = W1T + ((size_t)fid * 32 + lane) * 16;
    } else {
        // W2 fragment: g = (e*DT2 + dtg)*SC2 + sc ; K = s, N = d
        const int g   = fid - F1TOT;
        const int sc  = g % SC2;
        const int dtg = (g / SC2) % DT2;
        const int e   = g / (SC2 * DT2);
        const float* src = W2 + ((size_t)e * NSMAX + sc * 32 + hsel * 16) * ND
                              + dtg * 16 + ncol;
        #pragma unroll
        for (int j = 0; j < 16; ++j) o.u[j] = f2bf(src[(size_t)j * ND]);
        dst = W2T + ((size_t)g * 32 + lane) * 16;
    }
    *(uint4*)(dst)     = o.q[0];
    *(uint4*)(dst + 8) = o.q[1];
}

// =====================================================================
// Kernel 1: router (logits -> top4 -> softmax -> per-expert lists),
//           residual out = x, and (optional) bf16 copy of x.
// =====================================================================
__global__ __launch_bounds__(256) void router_kernel(
    const float* __restrict__ x, const float* __restrict__ Wr,
    const float* __restrict__ br, float* __restrict__ out,
    int* __restrict__ counts, int* __restrict__ list_tok,
    float* __restrict__ list_w, unsigned short* __restrict__ xbf)
{
    __shared__ __attribute__((aligned(16))) float xt[4][ND];   // 16 KB
    __shared__ float lg[4][NE];

    const int tid = threadIdx.x;
    const int b0  = blockIdx.x * 4;

    for (int i = tid; i < 4 * ND; i += 256) {
        int r = i >> 10, d = i & (ND - 1);
        float v = x[(size_t)(b0 + r) * ND + d];
        xt[r][d] = v;
        out[(size_t)(b0 + r) * ND + d] = v;   // residual out = x
    }
    __syncthreads();

    // bf16 copy of x for conversion-free staging in the MLP kernel
    if (xbf) {
        for (int i = tid; i < 4 * ND / 2; i += 256) {
            int r = i >> 9, d2 = (i & 511) * 2;
            *(unsigned*)&xbf[(size_t)(b0 + r) * ND + d2] =
                pack2bf(xt[r][d2], xt[r][d2 + 1]);
        }
    }

    const int row = tid >> 6;
    const int e   = tid & 63;
    float acc = br[e];
    #pragma unroll 8
    for (int d = 0; d < ND; ++d)
        acc += xt[row][d] * Wr[d * NE + e];
    lg[row][e] = acc;
    __syncthreads();

    if (tid < 4) {
        int   bi[NK];
        float bv[NK];
        #pragma unroll
        for (int k = 0; k < NK; ++k) {
            float mx = -3.4e38f; int mi = 0;
            for (int j = 0; j < NE; ++j) {
                float v = lg[tid][j];
                if (v > mx) { mx = v; mi = j; }
            }
            bv[k] = mx; bi[k] = mi;
            lg[tid][mi] = -3.4e38f;
        }
        float s = 0.f, wv[NK];
        #pragma unroll
        for (int k = 0; k < NK; ++k) { wv[k] = __expf(bv[k] - bv[0]); s += wv[k]; }
        float inv = 1.f / s;
        #pragma unroll
        for (int k = 0; k < NK; ++k) {
            int ex  = bi[k];
            int pos = atomicAdd(&counts[ex], 1);
            list_tok[ex * NB + pos] = b0 + tid;
            list_w [ex * NB + pos] = wv[k] * inv;
        }
    }
}

// =====================================================================
// Kernel 2: fused two-layer expert MLP on routed tokens, bf16 WMMA.
// grid = (NCHUNK, NE), 256 threads = 8 waves.
// Outer GEMM loops kept ROLLED (#pragma unroll 1) so only one group of
// 4 B-fragments (32 VGPRs) is in flight -> no scratch spills; within a
// group the 8 global b128 loads are issued ahead of the 4 WMMAs.
// =====================================================================
template<bool PRECONV>
__global__ __launch_bounds__(256) void moe_mlp_kernel(
    const float* __restrict__ x, const unsigned short* __restrict__ xbf,
    const float* __restrict__ W1, const float* __restrict__ b1,
    const float* __restrict__ g1, const float* __restrict__ be1,
    const float* __restrict__ m1, const float* __restrict__ v1,
    const float* __restrict__ W2, const float* __restrict__ b2,
    const float* __restrict__ g2, const float* __restrict__ be2,
    const float* __restrict__ m2, const float* __restrict__ v2,
    const unsigned short* __restrict__ W1T,
    const unsigned short* __restrict__ W2T,
    const int* __restrict__ counts, const int* __restrict__ list_tok,
    const float* __restrict__ list_w, float* __restrict__ out)
{
    __shared__ __attribute__((aligned(16))) unsigned short xs[16][ND];     // 32 KB
    __shared__ __attribute__((aligned(16))) unsigned short hs[16][SCHUNK]; //  4 KB
    __shared__ int   toks[16];
    __shared__ float wts[16];

    const int c    = blockIdx.x;       // s-chunk 0..11
    const int e    = blockIdx.y;       // expert
    const int tid  = threadIdx.x;
    const int lane = tid & 31;
    const int w    = tid >> 5;         // wave 0..7 (wave32)
    const int n    = counts[e];
    if (n == 0) return;                // uniform exit

    const int ncol = lane & 15;
    const int hsel = lane >> 4;

    // layer-1 folded BN for this wave's s-column
    const int s_idx = c * SCHUNK + w * 16 + ncol;
    const float a1 = g1[e * NSMAX + s_idx] * rsqrtf(v1[e * NSMAX + s_idx] + EPS);
    const float c1 = (b1[e * NSMAX + s_idx] - m1[e * NSMAX + s_idx]) * a1
                     + be1[e * NSMAX + s_idx];

    // layer-2 folded BN for this wave's 8 d-tiles
    float a2r[8], c2r[8];
    #pragma unroll
    for (int dt = 0; dt < 8; ++dt) {
        int d = w * 128 + dt * 16 + ncol;
        float a2 = g2[e * ND + d] * rsqrtf(v2[e * ND + d] + EPS);
        a2r[dt] = a2;
        c2r[dt] = (b2[e * ND + d] - m2[e * ND + d]) * a2 + be2[e * ND + d];
    }

    const size_t w1base = (size_t)e * ND * NSMAX;
    const size_t w2base = (size_t)e * NSMAX * ND;
    const int    st1    = c * 8 + w;   // layer-1 s-tile index
    // pre-swizzled fragment bases (halves): fragment = 512 halves (1 KB)
    const unsigned short* w1frag =
        W1T + ((size_t)(e * ST1 + st1) * KC1) * 512 + (size_t)lane * 16;
    const unsigned short* w2frag =
        W2T + (((size_t)(e * DT2 + w * 8)) * SC2 + c * 4) * 512 + (size_t)lane * 16;

    for (int t0 = 0; t0 < n; t0 += 16) {
        if (tid < 16) {
            int idx = t0 + tid;
            toks[tid] = (idx < n) ? list_tok[e * NB + idx] : -1;
            wts[tid]  = (idx < n) ? list_w [e * NB + idx] : 0.f;
        }
        __syncthreads();

        // ---- stage x tile (16 x 1024) ----
        if (PRECONV) {
            // bf16 source: 16 rows x 2 KB = 2048 x 16B chunks, 8 per thread
            #pragma unroll
            for (int q = 0; q < 8; ++q) {
                int chunk = q * 256 + tid;
                int m   = chunk >> 7;             // 128 chunks per row
                int col = (chunk & 127) * 8;      // halves
                int t = toks[m]; if (t < 0) t = toks[0];   // dummy rows: any valid row
                const unsigned short* gsrc = xbf + (size_t)t * ND + col;
#if USE_ASYNC_LDS
                __builtin_amdgcn_global_load_async_to_lds_b128(
                    (as1_v4i)(void*)gsrc, (as3_v4i)(void*)&xs[m][col], 0, 0);
#else
                *(uint4*)&xs[m][col] = *(const uint4*)gsrc;
#endif
            }
            wait_async_lds();
        } else {
            #pragma unroll 4
            for (int m = 0; m < 16; ++m) {
                int t = toks[m];
                int d4 = tid * 4;
                float4 v = make_float4(0.f, 0.f, 0.f, 0.f);
                if (t >= 0) v = *(const float4*)&x[(size_t)t * ND + d4];
                uint2 p; p.x = pack2bf(v.x, v.y); p.y = pack2bf(v.z, v.w);
                *(uint2*)&xs[m][d4] = p;
            }
        }
        __syncthreads();

        // ---- layer 1: acc = X(16xD) @ W1[:, s-tile]; groups of 4 WMMAs ----
        v8f acc = {0.f, 0.f, 0.f, 0.f, 0.f, 0.f, 0.f, 0.f};
        #pragma unroll 1
        for (int kc4 = 0; kc4 < KC1; kc4 += 4) {
            Frag Bf[4];
            if (PRECONV) {
                #pragma unroll
                for (int u = 0; u < 4; ++u) {
                    const u16x8* bp = (const u16x8*)(w1frag + (size_t)(kc4 + u) * 512);
                    Bf[u].h[0] = bp[0]; Bf[u].h[1] = bp[1];
                }
            } else {
                #pragma unroll
                for (int u = 0; u < 4; ++u) {
                    const int kb = (kc4 + u) * 32;
                    const float* wp = W1 + w1base
                                    + (size_t)(kb + hsel * 16) * NSMAX + s_idx;
                    #pragma unroll
                    for (int j = 0; j < 16; ++j) Bf[u].u[j] = f2bf(wp[(size_t)j * NSMAX]);
                }
            }
            #pragma unroll
            for (int u = 0; u < 4; ++u) {
                const int kb = (kc4 + u) * 32;
                Frag A;
                A.h[0] = *(const u16x8*)&xs[ncol][kb + hsel * 16];
                A.h[1] = *(const u16x8*)&xs[ncol][kb + hsel * 16 + 8];
                acc = __builtin_amdgcn_wmma_f32_16x16x32_bf16(
                    false, A.v, false, Bf[u].v, (short)0, acc, false, false);
            }
        }
        // BN1 + ReLU -> H tile
        #pragma unroll
        for (int r = 0; r < 8; ++r) {
            float hv = fmaxf(acc[r] * a1 + c1, 0.f);
            hs[r + hsel * 8][w * 16 + ncol] = f2bf(hv);
        }
        __syncthreads();

        // ---- layer 2: Y[16 x 128] = H(16x128) @ W2[s-chunk, d-range] ----
        #pragma unroll 1
        for (int dt = 0; dt < 8; ++dt) {
            Frag B2f[4];
            if (PRECONV) {
                #pragma unroll
                for (int u = 0; u < 4; ++u) {
                    const u16x8* bp =
                        (const u16x8*)(w2frag + ((size_t)dt * SC2 + u) * 512);
                    B2f[u].h[0] = bp[0]; B2f[u].h[1] = bp[1];
                }
            } else {
                #pragma unroll
                for (int u = 0; u < 4; ++u) {
                    const int kb2 = u * 32;
                    const float* wp2 = W2 + w2base
                                     + (size_t)(c * SCHUNK + kb2 + hsel * 16) * ND
                                     + (w * 128 + dt * 16 + ncol);
                    #pragma unroll
                    for (int j = 0; j < 16; ++j) B2f[u].u[j] = f2bf(wp2[(size_t)j * ND]);
                }
            }
            v8f acc2 = {0.f, 0.f, 0.f, 0.f, 0.f, 0.f, 0.f, 0.f};
            #pragma unroll
            for (int u = 0; u < 4; ++u) {
                const int kb2 = u * 32;
                Frag A2;
                A2.h[0] = *(const u16x8*)&hs[ncol][kb2 + hsel * 16];
                A2.h[1] = *(const u16x8*)&hs[ncol][kb2 + hsel * 16 + 8];
                acc2 = __builtin_amdgcn_wmma_f32_16x16x32_bf16(
                    false, A2.v, false, B2f[u].v, (short)0, acc2, false, false);
            }
            const int d = w * 128 + dt * 16 + ncol;
            #pragma unroll
            for (int r = 0; r < 8; ++r) {
                int m = r + hsel * 8;
                int t = toks[m];
                if (t >= 0) {
                    float val = acc2[r] * a2r[dt];
                    if (c == 0) val += c2r[dt];
                    atomicAdd(&out[(size_t)t * ND + d], wts[m] * val);
                }
            }
        }
        __syncthreads();
    }
}

// =====================================================================
// Host launcher.
// ws layout: counts(1KB) | list_tok(512KB) | list_w(512KB)
//            | @2MB: xbf bf16 (4MB) | @8MB: W1T (192MB) | W2T (192MB)
// =====================================================================
extern "C" void kernel_launch(void* const* d_in, const int* in_sizes, int n_in,
                              void* d_out, int out_size, void* d_ws, size_t ws_size,
                              hipStream_t stream)
{
    const float* x   = (const float*)d_in[0];
    const float* W1  = (const float*)d_in[1];
    const float* b1  = (const float*)d_in[2];
    const float* g1  = (const float*)d_in[3];
    const float* be1 = (const float*)d_in[4];
    const float* m1  = (const float*)d_in[5];
    const float* v1  = (const float*)d_in[6];
    const float* W2  = (const float*)d_in[7];
    const float* b2  = (const float*)d_in[8];
    const float* g2  = (const float*)d_in[9];
    const float* be2 = (const float*)d_in[10];
    const float* m2  = (const float*)d_in[11];
    const float* v2  = (const float*)d_in[12];
    const float* Wr  = (const float*)d_in[13];
    const float* br  = (const float*)d_in[14];
    float* out = (float*)d_out;

    char* ws = (char*)d_ws;
    int*   counts   = (int*)ws;
    int*   list_tok = (int*)(ws + 1024);
    float* list_w   = (float*)(ws + 1024 + (size_t)NE * NB * sizeof(int));

    const size_t xbf_off   = 2u * 1024u * 1024u;
    const size_t wbf_off   = 8u * 1024u * 1024u;
    const size_t w1t_bytes = (size_t)F1TOT * 1024;                // 192 MB
    const size_t w2t_bytes = (size_t)F2TOT * 1024;                // 192 MB
    unsigned short* xbf = (unsigned short*)(ws + xbf_off);
    unsigned short* W1T = (unsigned short*)(ws + wbf_off);
    unsigned short* W2T = (unsigned short*)(ws + wbf_off + w1t_bytes);
    const bool preconv = ws_size >= wbf_off + w1t_bytes + w2t_bytes;

    (void)hipMemsetAsync(counts, 0, NE * sizeof(int), stream);

    router_kernel<<<NB / 4, 256, 0, stream>>>(x, Wr, br, out, counts,
                                              list_tok, list_w,
                                              preconv ? xbf : nullptr);

    dim3 grid(NCHUNK, NE);
    if (preconv) {
        convert_weights_kernel<<<(F1TOT + F2TOT) / 8, 256, 0, stream>>>(W1, W2, W1T, W2T);
        moe_mlp_kernel<true><<<grid, 256, 0, stream>>>(
            x, xbf, W1, b1, g1, be1, m1, v1, W2, b2, g2, be2, m2, v2,
            W1T, W2T, counts, list_tok, list_w, out);
    } else {
        moe_mlp_kernel<false><<<grid, 256, 0, stream>>>(
            x, nullptr, W1, b1, g1, be1, m1, v1, W2, b2, g2, be2, m2, v2,
            nullptr, nullptr, counts, list_tok, list_w, out);
    }
}
